// SharedSubspaceEncoderLayer_24120536334978
// MI455X (gfx1250) — matrix operand order repaired and checked by reading, not compiled, not run on hardware
//
#include <hip/hip_runtime.h>

// ---------------------------------------------------------------------------
// MI455X (gfx1250) implementation of SharedSubspaceEncoderLayer.
// - All matmuls (GEMM chain + both attention GEMMs) use
//   v_wmma_f32_16x16x32_bf16 (wave32 WMMA, fp32 accumulate).
// - Global->LDS staging uses GLOBAL_LOAD_ASYNC_TO_LDS_B128 (ASYNCcnt) when
//   the builtin is available, with double-buffered software pipelining in
//   the GEMM so DMA overlaps WMMA.
// ---------------------------------------------------------------------------

#define DEV __device__ __forceinline__

typedef __bf16 bf16_t;
typedef __attribute__((ext_vector_type(16))) __bf16 bf16x16;
typedef __attribute__((ext_vector_type(8)))  float  f32x8;

constexpr int Bc  = 4;
constexpr int Tc  = 2048;
constexpr int Hc  = 16;
constexpr int Dhc = 64;
constexpr int NTOK = Bc * Tc;   // 8192 tokens

// ---- async global->LDS support (guarded; falls back to sync copies) -------
#if defined(__gfx1250__) && __has_builtin(__builtin_amdgcn_global_load_async_to_lds_b128)
#define HAS_ASYNC_LDS 1
typedef int v4i __attribute__((vector_size(16)));   // matches builtin param type
typedef __attribute__((address_space(1))) v4i* g_v4i_p;   // global (AS1)
typedef __attribute__((address_space(3))) v4i* l_v4i_p;   // LDS (AS3)
DEV g_v4i_p to_g(const void* p) { return (g_v4i_p)(unsigned long long)p; }
// flat LDS address: low 32 bits are the LDS byte offset (ISA aperture rule)
DEV l_v4i_p to_l(void* p) { return (l_v4i_p)(unsigned)(unsigned long long)p; }
DEV void wait_async_lds() {
#if __has_builtin(__builtin_amdgcn_s_wait_asynccnt)
  __builtin_amdgcn_s_wait_asynccnt(0);
#else
  asm volatile("s_wait_asynccnt 0x0" ::: "memory");
#endif
}
#else
#define HAS_ASYNC_LDS 0
DEV void wait_async_lds() {}
#endif

DEV bf16_t f2bf(float f) {
  union { float f; unsigned u; } v; v.f = f;
  unsigned r = v.u + 0x7fffu + ((v.u >> 16) & 1u);   // round-to-nearest-even
  unsigned short h = (unsigned short)(r >> 16);
  bf16_t b;
  __builtin_memcpy(&b, &h, sizeof(b));
  return b;
}

DEV f32x8 wmma_bf16(bf16x16 a, bf16x16 b, f32x8 c) {
  // (neg_a, A, neg_b, B, c_mod, C, reuse_a, reuse_b)
  return __builtin_amdgcn_wmma_f32_16x16x32_bf16(false, a, false, b,
                                                 (short)0, c, false, false);
}

// ---------------------------------------------------------------------------
// Elementwise helpers
// ---------------------------------------------------------------------------
__global__ void f32_to_bf16_kernel(const float* __restrict__ in,
                                   bf16_t* __restrict__ out, long n) {
  long i = (long)blockIdx.x * blockDim.x + threadIdx.x;
  if (i < n) out[i] = f2bf(in[i]);
}

__global__ void gelu_bf16_kernel(const float* __restrict__ in,
                                 bf16_t* __restrict__ out, long n) {
  long i = (long)blockIdx.x * blockDim.x + threadIdx.x;
  if (i < n) {
    float x = in[i];
    float t = tanhf(0.7978845608028654f * (x + 0.044715f * x * x * x));
    out[i] = f2bf(0.5f * x * (1.0f + t));
  }
}

__global__ void add_f32_kernel(const float* __restrict__ a,
                               const float* __restrict__ b,
                               float* __restrict__ o, long n) {
  long i = (long)blockIdx.x * blockDim.x + threadIdx.x;
  if (i < n) o[i] = a[i] + b[i];
}

// ---------------------------------------------------------------------------
// RMSNorm (fp32 in, bf16 out), one 256-thread block (8 waves) per row.
// ---------------------------------------------------------------------------
__global__ __launch_bounds__(256) void rmsnorm_bf16_kernel(
    const float* __restrict__ x, const float* __restrict__ w,
    bf16_t* __restrict__ out, int D) {
  __shared__ float red[8];
  long row = blockIdx.x;
  const float* xr = x + row * D;
  float s = 0.f;
  for (int i = threadIdx.x; i < D; i += 256) { float v = xr[i]; s += v * v; }
  #pragma unroll
  for (int off = 16; off > 0; off >>= 1) s += __shfl_xor(s, off, 32);
  if ((threadIdx.x & 31) == 0) red[threadIdx.x >> 5] = s;
  __syncthreads();
  float tot = 0.f;
  #pragma unroll
  for (int i = 0; i < 8; i++) tot += red[i];
  float rinv = rsqrtf(tot / (float)D + 1e-6f);
  bf16_t* orow = out + row * D;
  for (int i = threadIdx.x; i < D; i += 256) orow[i] = f2bf(xr[i] * rinv * w[i]);
}

// ---------------------------------------------------------------------------
// Tiled bf16 WMMA GEMM: C(f32, MxN) = A(bf16, MxK, lda) * B(bf16, KxN, ldb)
// Block = 256 threads (8 waves), macro tile 128x64, BK = 32.
// Wave grid 4(M) x 2(N); each wave owns a 32x32 sub-tile = 4 WMMA tiles.
// Double-buffered LDS staging: tile i+1 streams in via async global->LDS
// while tile i feeds the WMMAs.  Requires M%128==0, N%64==0, K%32==0.
// ---------------------------------------------------------------------------
DEV void stage_gemm_tile(const bf16_t* __restrict__ A, long lda,
                         const bf16_t* __restrict__ B, long ldb,
                         long bm, long bn, int k0, int tid,
                         bf16_t (*__restrict__ As)[32],
                         bf16_t (*__restrict__ Bs)[64]) {
  int ra = tid >> 1, ca = (tid & 1) * 16;                 // A: 128x32
  const bf16_t* srcA = A + (bm + ra) * lda + k0 + ca;
  int rb = tid >> 3, cb = (tid & 7) * 8;                  // B: 32x64
  const bf16_t* srcB = B + (long)(k0 + rb) * ldb + bn + cb;
#if HAS_ASYNC_LDS
  __builtin_amdgcn_global_load_async_to_lds_b128(to_g(srcA),     to_l(&As[ra][ca]),     0, 0);
  __builtin_amdgcn_global_load_async_to_lds_b128(to_g(srcA + 8), to_l(&As[ra][ca + 8]), 0, 0);
  __builtin_amdgcn_global_load_async_to_lds_b128(to_g(srcB),     to_l(&Bs[rb][cb]),     0, 0);
#else
  #pragma unroll
  for (int j = 0; j < 16; j++) As[ra][ca + j] = srcA[j];
  #pragma unroll
  for (int j = 0; j < 8; j++) Bs[rb][cb + j] = srcB[j];
#endif
  __builtin_prefetch(srcA + 32, 0, 1);                    // speculative, next+1
  __builtin_prefetch(srcB + 32 * ldb, 0, 1);
}

__global__ __launch_bounds__(256) void gemm_bf16_kernel(
    const bf16_t* __restrict__ A, long lda,
    const bf16_t* __restrict__ B, long ldb,
    float* __restrict__ C, long ldc, int K) {
  __shared__ bf16_t As[2][128][32];   // 2 x 8 KB
  __shared__ bf16_t Bs[2][32][64];    // 2 x 4 KB
  const long bm = (long)blockIdx.y * 128;
  const long bn = (long)blockIdx.x * 64;
  const int tid  = threadIdx.x;
  const int wave = tid >> 5, lane = tid & 31;
  const int hf = lane >> 4, lm = lane & 15;
  const int wm = wave >> 1, wn = wave & 1;

  f32x8 acc[2][2];
  #pragma unroll
  for (int i = 0; i < 2; i++)
    #pragma unroll
    for (int j = 0; j < 2; j++)
      #pragma unroll
      for (int e = 0; e < 8; e++) acc[i][j][e] = 0.f;

  // prologue: stage tile 0
  stage_gemm_tile(A, lda, B, ldb, bm, bn, 0, tid, As[0], Bs[0]);
  wait_async_lds();
  __syncthreads();

  for (int k0 = 0; k0 < K; k0 += 32) {
    const int cur = (k0 >> 5) & 1;
    if (k0 + 32 < K)   // kick off DMA for next tile into the other buffer
      stage_gemm_tile(A, lda, B, ldb, bm, bn, k0 + 32, tid,
                      As[cur ^ 1], Bs[cur ^ 1]);

    bf16_t (*Ap)[32] = As[cur];
    bf16_t (*Bp)[64] = Bs[cur];

    // A fragment: lane holds row M=lane%16; K = (e&7) + 8*half + 16*(e>>3)
    bf16x16 af[2], bfb[2];
    #pragma unroll
    for (int i = 0; i < 2; i++) {
      int m = wm * 32 + i * 16 + lm;
      #pragma unroll
      for (int e = 0; e < 16; e++)
        af[i][e] = Ap[m][(e & 7) + hf * 8 + (e >> 3) * 16];
    }
    // B fragment: lane holds column N=lane%16; K = (e&7) + 8*half + 16*(e>>3)
    #pragma unroll
    for (int j = 0; j < 2; j++) {
      int nn = wn * 32 + j * 16 + lm;
      #pragma unroll
      for (int e = 0; e < 16; e++)
        bfb[j][e] = Bp[(e & 7) + hf * 8 + (e >> 3) * 16][nn];
    }
    #pragma unroll
    for (int i = 0; i < 2; i++)
      #pragma unroll
      for (int j = 0; j < 2; j++)
        acc[i][j] = wmma_bf16(af[i], bfb[j], acc[i][j]);

    wait_async_lds();     // next tile resident in LDS
    __syncthreads();      // all waves done with cur buffer + see new tile
  }

  // C/D layout: lane -> column N=lane%16; VGPR r -> row r + 8*(lane/16)
  #pragma unroll
  for (int i = 0; i < 2; i++)
    #pragma unroll
    for (int j = 0; j < 2; j++) {
      long n = bn + wn * 32 + j * 16 + lm;
      #pragma unroll
      for (int r = 0; r < 8; r++) {
        long m = bm + wm * 32 + i * 16 + r + hf * 8;
        C[m * ldc + n] = acc[i][j][r];
      }
    }
}

// ---------------------------------------------------------------------------
// RoPE + transpose to (B,H,T,Dh) bf16.
// q input layout: (B,T,H*Dh) fp32.  kv input layout: (B,T,H*2*Dh) fp32.
// rotate_half on first 32 dims: out[d<16] = x*cos - x[d+16]*sin
//                               out[16<=d<32] = x*cos + x[d-16]*sin
// ---------------------------------------------------------------------------
__global__ void rope_q_kernel(const float* __restrict__ q,
                              const float* __restrict__ cosT,
                              const float* __restrict__ sinT,
                              bf16_t* __restrict__ qout) {
  long i = (long)blockIdx.x * blockDim.x + threadIdx.x;
  if (i >= (long)NTOK * Hc * Dhc) return;
  int d = (int)(i & 63);
  long r = i >> 6;
  int h = (int)(r % Hc);
  long bt = r / Hc;
  int t = (int)(bt % Tc);
  int b = (int)(bt / Tc);
  float v = q[i];
  float outv = v;
  if (d < 32) {
    float c = cosT[t * 32 + d], s = sinT[t * 32 + d];
    outv = (d < 16) ? (v * c - q[i + 16] * s) : (v * c + q[i - 16] * s);
  }
  qout[(((size_t)b * Hc + h) * Tc + t) * Dhc + d] = f2bf(outv);
}

__global__ void rope_kv_kernel(const float* __restrict__ kv,
                               const float* __restrict__ cosT,
                               const float* __restrict__ sinT,
                               bf16_t* __restrict__ kout,
                               bf16_t* __restrict__ vout) {
  long i = (long)blockIdx.x * blockDim.x + threadIdx.x;
  if (i >= (long)NTOK * Hc * Dhc) return;
  int d = (int)(i & 63);
  long r = i >> 6;
  int h = (int)(r % Hc);
  long bt = r / Hc;
  int t = (int)(bt % Tc);
  int b = (int)(bt / Tc);
  size_t src = (((size_t)b * Tc + t) * Hc + h) * (2 * Dhc);
  float kvv = kv[src + d];
  float ko = kvv;
  if (d < 32) {
    float c = cosT[t * 32 + d], s = sinT[t * 32 + d];
    ko = (d < 16) ? (kvv * c - kv[src + d + 16] * s)
                  : (kvv * c + kv[src + d - 16] * s);
  }
  size_t dst = (((size_t)b * Hc + h) * Tc + t) * Dhc + d;
  kout[dst] = f2bf(ko);
  vout[dst] = f2bf(kv[src + Dhc + d]);
}

// ---------------------------------------------------------------------------
// Flash attention (full, non-causal).  Q/K/V: (B,H,T,64) bf16.
// Block = 8 waves; each wave owns 16 q-rows (block tile = 128 q rows).
// Per iteration: 32 keys staged in LDS.  V streams in via async global->LDS;
// K is staged transposed (so the S-GEMM B-fragment reads are contiguous).
// Online softmax with shfl_xor row reductions inside 16-lane groups.
// P converted C-layout -> A-layout through per-wave LDS scratch.
// ctx out: (B,T,H*64) bf16.
// ---------------------------------------------------------------------------
__global__ __launch_bounds__(256) void attention_kernel(
    const bf16_t* __restrict__ Q, const bf16_t* __restrict__ K,
    const bf16_t* __restrict__ V, bf16_t* __restrict__ ctx) {
  __shared__ bf16_t Ks[64][32];     // K^T tile: [d][key]
  __shared__ bf16_t Vs[32][64];     // V tile:   [key][d]
  __shared__ bf16_t Ps[8][16][32];  // per-wave P scratch (C->A relayout)

  const int bh = blockIdx.x;
  const int bb = bh / Hc, hh = bh % Hc;
  const int q0 = blockIdx.y * 128;
  const int tid  = threadIdx.x;
  const int wave = tid >> 5, lane = tid & 31;
  const int hf = lane >> 4, lm = lane & 15;

  const bf16_t* Qb = Q + (size_t)bh * Tc * Dhc;
  const bf16_t* Kb = K + (size_t)bh * Tc * Dhc;
  const bf16_t* Vb = V + (size_t)bh * Tc * Dhc;

  // Q fragments (A layout) for this wave's 16 rows; resident all kernel.
  bf16x16 qf[2];
  {
    const bf16_t* qr = Qb + (size_t)(q0 + wave * 16 + lm) * Dhc;
    #pragma unroll
    for (int c = 0; c < 2; c++)
      #pragma unroll
      for (int e = 0; e < 16; e++)
        qf[c][e] = qr[c * 32 + (e & 7) + hf * 8 + (e >> 3) * 16];
  }

  f32x8 o[4];
  #pragma unroll
  for (int c = 0; c < 4; c++)
    #pragma unroll
    for (int e = 0; e < 8; e++) o[c][e] = 0.f;
  float mrow[8], lrow[8];
  #pragma unroll
  for (int r = 0; r < 8; r++) { mrow[r] = -1e30f; lrow[r] = 0.f; }

  for (int kt = 0; kt < Tc; kt += 32) {
    __syncthreads();
    {   // stage K (transposed, via VGPRs) and V (async DMA) tiles
      int key = tid >> 3;
      int db  = (tid & 7) * 8;
      const bf16_t* ks = Kb + (size_t)(kt + key) * Dhc + db;
      const bf16_t* vs = Vb + (size_t)(kt + key) * Dhc + db;
      #pragma unroll
      for (int j = 0; j < 8; j++) Ks[db + j][key] = ks[j];
#if HAS_ASYNC_LDS
      __builtin_amdgcn_global_load_async_to_lds_b128(to_g(vs), to_l(&Vs[key][db]), 0, 0);
#else
      #pragma unroll
      for (int j = 0; j < 8; j++) Vs[key][db + j] = vs[j];
#endif
    }
    wait_async_lds();
    __syncthreads();

    // S = Q K^T  (16q x 32k), accumulated over two 32-dim chunks
    f32x8 s0, s1;
    #pragma unroll
    for (int e = 0; e < 8; e++) { s0[e] = 0.f; s1[e] = 0.f; }
    #pragma unroll
    for (int c = 0; c < 2; c++) {
      int d = c * 32 + hf * 8;   // reduction base for B fragment (table order)
      bf16x16 b0, b1;
      #pragma unroll
      for (int e = 0; e < 16; e++) b0[e] = Ks[d + (e & 7) + (e >> 3) * 16][lm];
      #pragma unroll
      for (int e = 0; e < 16; e++) b1[e] = Ks[d + (e & 7) + (e >> 3) * 16][16 + lm];
      s0 = wmma_bf16(qf[c], b0, s0);
      s1 = wmma_bf16(qf[c], b1, s1);
    }
    #pragma unroll
    for (int e = 0; e < 8; e++) { s0[e] *= 0.125f; s1[e] *= 0.125f; }

    // Online softmax update (rows live in 16-lane groups)
    #pragma unroll
    for (int r = 0; r < 8; r++) {
      float mx = fmaxf(s0[r], s1[r]);
      #pragma unroll
      for (int off = 1; off < 16; off <<= 1)
        mx = fmaxf(mx, __shfl_xor(mx, off, 32));
      float mnew = fmaxf(mrow[r], mx);
      float sc = __expf(mrow[r] - mnew);
      mrow[r] = mnew;
      float p0 = __expf(s0[r] - mnew);
      float p1 = __expf(s1[r] - mnew);
      s0[r] = p0; s1[r] = p1;
      float rs = p0 + p1;
      #pragma unroll
      for (int off = 1; off < 16; off <<= 1) rs += __shfl_xor(rs, off, 32);
      lrow[r] = lrow[r] * sc + rs;
      #pragma unroll
      for (int c = 0; c < 4; c++) o[c][r] *= sc;
    }

    // P: C layout -> row-major LDS scratch (own wave region)
    #pragma unroll
    for (int r = 0; r < 8; r++) {
      Ps[wave][r + hf * 8][lm]      = f2bf(s0[r]);
      Ps[wave][r + hf * 8][16 + lm] = f2bf(s1[r]);
    }
    __syncthreads();

    // P fragment (A layout) and O += P * V
    bf16x16 pf;
    #pragma unroll
    for (int e = 0; e < 16; e++)
      pf[e] = Ps[wave][lm][(e & 7) + hf * 8 + (e >> 3) * 16];
    #pragma unroll
    for (int c = 0; c < 4; c++) {
      bf16x16 vf;
      int rr = hf * 8;   // key (reduction) base, table order
      #pragma unroll
      for (int e = 0; e < 16; e++) vf[e] = Vs[rr + (e & 7) + (e >> 3) * 16][c * 16 + lm];
      o[c] = wmma_bf16(pf, vf, o[c]);
    }
  }

  // normalize + store ctx (B,T,H*64)
  #pragma unroll
  for (int c = 0; c < 4; c++)
    #pragma unroll
    for (int r = 0; r < 8; r++) {
      int m = q0 + wave * 16 + r + hf * 8;
      int d = c * 16 + lm;
      float val = o[c][r] / lrow[r];
      ctx[(((size_t)bb * Tc + m) * Hc + hh) * Dhc + d] = f2bf(val);
    }
}

// ---------------------------------------------------------------------------
// Host orchestration
// ---------------------------------------------------------------------------
extern "C" void kernel_launch(void* const* d_in, const int* in_sizes, int n_in,
                              void* d_out, int out_size, void* d_ws, size_t ws_size,
                              hipStream_t stream) {
  (void)in_sizes; (void)n_in; (void)out_size; (void)ws_size;
  const float* x        = (const float*)d_in[0];
  const float* ln1_w    = (const float*)d_in[1];
  const float* a_ln_w   = (const float*)d_in[2];
  const float* qkv_a_w  = (const float*)d_in[3];
  const float* q_b_w    = (const float*)d_in[4];
  const float* kv_b_w   = (const float*)d_in[5];
  const float* o_a_w    = (const float*)d_in[6];
  const float* o_b_w    = (const float*)d_in[7];
  const float* ln2_w    = (const float*)d_in[8];
  const float* ffn_in_a = (const float*)d_in[9];
  const float* ffn_in_b = (const float*)d_in[10];
  const float* ffn_out_a= (const float*)d_in[11];
  const float* ffn_out_b= (const float*)d_in[12];
  const float* cosT     = (const float*)d_in[13];
  const float* sinT     = (const float*)d_in[14];
  float* out = (float*)d_out;

  char* ws = (char*)d_ws;
  size_t off = 0;
  auto alloc = [&](size_t bytes) -> void* {
    void* p = ws + off;
    off += (bytes + 255) & ~(size_t)255;
    return p;
  };
  auto cvt = [&](const float* src, size_t n) -> bf16_t* {
    bf16_t* dst = (bf16_t*)alloc(n * sizeof(bf16_t));
    f32_to_bf16_kernel<<<dim3((unsigned)((n + 255) / 256)), 256, 0, stream>>>(src, dst, (long)n);
    return dst;
  };
  auto gemm = [&](const bf16_t* A, long lda, const bf16_t* B, long ldb,
                  float* C, long ldc, int M, int N, int K) {
    dim3 g((unsigned)(N / 64), (unsigned)(M / 128));
    gemm_bf16_kernel<<<g, 256, 0, stream>>>(A, lda, B, ldb, C, ldc, K);
  };

  // --- weights -> bf16 -----------------------------------------------------
  bf16_t* wqkv = cvt(qkv_a_w, (size_t)1024 * 768);
  bf16_t* wqb  = cvt(q_b_w,   (size_t)512 * 1024);
  bf16_t* wkvb = cvt(kv_b_w,  (size_t)256 * 2048);
  bf16_t* woa  = cvt(o_a_w,   (size_t)1024 * 512);
  bf16_t* wob  = cvt(o_b_w,   (size_t)512 * 1024);
  bf16_t* wfia = cvt(ffn_in_a, (size_t)1024 * 384);
  bf16_t* wfib = cvt(ffn_in_b, (size_t)384 * 4096);
  bf16_t* wfoa = cvt(ffn_out_a,(size_t)4096 * 384);
  bf16_t* wfob = cvt(ffn_out_b,(size_t)384 * 1024);

  // --- attention path ------------------------------------------------------
  bf16_t* h_bf = (bf16_t*)alloc((size_t)NTOK * 1024 * 2);
  rmsnorm_bf16_kernel<<<NTOK, 256, 0, stream>>>(x, ln1_w, h_bf, 1024);

  float* lat_f = (float*)alloc((size_t)NTOK * 768 * 4);
  gemm(h_bf, 1024, wqkv, 768, lat_f, 768, NTOK, 768, 1024);

  bf16_t* lat_bf = (bf16_t*)alloc((size_t)NTOK * 768 * 2);
  rmsnorm_bf16_kernel<<<NTOK, 256, 0, stream>>>(lat_f, a_ln_w, lat_bf, 768);

  float* q_f  = (float*)alloc((size_t)NTOK * 1024 * 4);
  gemm(lat_bf, 768, wqb, 1024, q_f, 1024, NTOK, 1024, 512);
  float* kv_f = (float*)alloc((size_t)NTOK * 2048 * 4);
  gemm(lat_bf + 512, 768, wkvb, 2048, kv_f, 2048, NTOK, 2048, 256);

  bf16_t* q_bf = (bf16_t*)alloc((size_t)NTOK * 1024 * 2);
  bf16_t* k_bf = (bf16_t*)alloc((size_t)NTOK * 1024 * 2);
  bf16_t* v_bf = (bf16_t*)alloc((size_t)NTOK * 1024 * 2);
  long nq = (long)NTOK * Hc * Dhc;
  rope_q_kernel<<<dim3((unsigned)((nq + 255) / 256)), 256, 0, stream>>>(q_f, cosT, sinT, q_bf);
  rope_kv_kernel<<<dim3((unsigned)((nq + 255) / 256)), 256, 0, stream>>>(kv_f, cosT, sinT, k_bf, v_bf);

  bf16_t* ctx_bf = (bf16_t*)alloc((size_t)NTOK * 1024 * 2);
  attention_kernel<<<dim3(Bc * Hc, Tc / 128), 256, 0, stream>>>(q_bf, k_bf, v_bf, ctx_bf);

  float* t1_f = (float*)alloc((size_t)NTOK * 512 * 4);
  gemm(ctx_bf, 1024, woa, 512, t1_f, 512, NTOK, 512, 1024);
  bf16_t* t1_bf = (bf16_t*)alloc((size_t)NTOK * 512 * 2);
  f32_to_bf16_kernel<<<dim3((unsigned)(((long)NTOK * 512 + 255) / 256)), 256, 0, stream>>>(
      t1_f, t1_bf, (long)NTOK * 512);
  float* attn_f = (float*)alloc((size_t)NTOK * 1024 * 4);
  gemm(t1_bf, 512, wob, 1024, attn_f, 1024, NTOK, 1024, 512);

  float* x1 = (float*)alloc((size_t)NTOK * 1024 * 4);
  add_f32_kernel<<<dim3((unsigned)(((long)NTOK * 1024 + 255) / 256)), 256, 0, stream>>>(
      x, attn_f, x1, (long)NTOK * 1024);

  // --- FFN path ------------------------------------------------------------
  bf16_t* h2_bf = (bf16_t*)alloc((size_t)NTOK * 1024 * 2);
  rmsnorm_bf16_kernel<<<NTOK, 256, 0, stream>>>(x1, ln2_w, h2_bf, 1024);

  float* f1_f = (float*)alloc((size_t)NTOK * 384 * 4);
  gemm(h2_bf, 1024, wfia, 384, f1_f, 384, NTOK, 384, 1024);
  bf16_t* f1_bf = (bf16_t*)alloc((size_t)NTOK * 384 * 2);
  f32_to_bf16_kernel<<<dim3((unsigned)(((long)NTOK * 384 + 255) / 256)), 256, 0, stream>>>(
      f1_f, f1_bf, (long)NTOK * 384);

  float* f2_f = (float*)alloc((size_t)NTOK * 4096 * 4);
  gemm(f1_bf, 384, wfib, 4096, f2_f, 4096, NTOK, 4096, 384);
  bf16_t* f2_bf = (bf16_t*)alloc((size_t)NTOK * 4096 * 2);
  gelu_bf16_kernel<<<dim3((unsigned)(((long)NTOK * 4096 + 255) / 256)), 256, 0, stream>>>(
      f2_f, f2_bf, (long)NTOK * 4096);

  float* f3_f = (float*)alloc((size_t)NTOK * 384 * 4);
  gemm(f2_bf, 4096, wfoa, 384, f3_f, 384, NTOK, 384, 4096);
  bf16_t* f3_bf = (bf16_t*)alloc((size_t)NTOK * 384 * 2);
  f32_to_bf16_kernel<<<dim3((unsigned)(((long)NTOK * 384 + 255) / 256)), 256, 0, stream>>>(
      f3_f, f3_bf, (long)NTOK * 384);

  float* f4_f = (float*)alloc((size_t)NTOK * 1024 * 4);
  gemm(f3_bf, 384, wfob, 1024, f4_f, 1024, NTOK, 1024, 384);

  add_f32_kernel<<<dim3((unsigned)(((long)NTOK * 1024 + 255) / 256)), 256, 0, stream>>>(
      x1, f4_f, out, (long)NTOK * 1024);
}
